// DigitConvolutionalModel_970662609453
// MI455X (gfx1250) — matrix operand order repaired and checked by reading, not compiled, not run on hardware
//
#include <hip/hip_runtime.h>
#include <hip/hip_bf16.h>
#include <stdint.h>

// ---------------- Types for CDNA5 WMMA ----------------
typedef __attribute__((ext_vector_type(16))) __bf16 v16bf;
typedef __attribute__((ext_vector_type(8)))  float  v8f;

union ABFrag {
    uint4 q[2];      // two 16-byte LDS loads
    v16bf v;
};

__device__ __forceinline__ uint16_t f2bf(float f) {
    uint32_t u = __float_as_uint(f);
    u += 0x7FFFu + ((u >> 16) & 1u);   // round-to-nearest-even
    return (uint16_t)(u >> 16);
}

// ---------------- Geometry ----------------
#define BATCH   65536
#define MTILE   64          // batch rows per workgroup
#define IMGW    28
#define CONVW   26
#define CONVOUT 676
#define K1P     704         // 676 padded to 22 * 32
#define KT1     22
#define HID     200
#define HP      224         // 200 padded to 14 * 16 (and 7 * 32)
#define KT2     7
#define OUTN    10
#define OUTP    16

// Shared memory (bytes): hA 64*704*2 + hB 64*224*2 + double Wbuf 2*224*32*2
#define SMEM_BYTES ((MTILE * K1P + MTILE * HP + 2 * HP * 32) * 2)

// ---------------- Weight / bias packing ----------------
// dst[(kt*NP + n)*32 + kk] = bf16(src[k*Ns + n]) with k = kt*32+kk, zero padded.
__global__ void pack_w_kernel(const float* __restrict__ src, int Ks, int Ns,
                              uint16_t* __restrict__ dst, int KT, int NP) {
    int total = KT * NP * 32;
    int i = blockIdx.x * blockDim.x + threadIdx.x;
    if (i >= total) return;
    int kk = i & 31;
    int t  = i >> 5;
    int n  = t % NP;
    int kt = t / NP;
    int k  = kt * 32 + kk;
    float v = (k < Ks && n < Ns) ? src[(size_t)k * Ns + n] : 0.0f;
    dst[i] = f2bf(v);
}

__global__ void pack_b_kernel(const float* __restrict__ src, int N,
                              float* __restrict__ dst, int NP) {
    int i = blockIdx.x * blockDim.x + threadIdx.x;
    if (i >= NP) return;
    dst[i] = (i < N) ? src[i] : 0.0f;
}

// ---------------- Fused GEMM stage (one layer) ----------------
// A: LDS bf16 [64][lda].  Weights: global bf16 packed [kt][NP][32].
// Double-buffered weight slices; next slice is prefetched into registers
// while the WMMA chain runs, then stored to the alternate LDS buffer.
template <int NP, int NT, bool RELU, bool TO_LDS>
__device__ __forceinline__ void gemm_tile(
    const uint16_t* __restrict__ A, int lda, int ktiles,
    const uint32_t* __restrict__ Wg32,
    uint16_t* __restrict__ Wlds,          // 2 * NP*32 bf16
    const float* __restrict__ bias,
    uint16_t* __restrict__ Olds,
    float* __restrict__ Og, int row0,
    int tid)
{
    const int lane = tid & 31;
    const int wave = tid >> 5;
    const int g    = lane >> 4;   // half-wave group
    const int ln   = lane & 15;
    const int mt   = wave & 3;    // m-tile (16 rows) owned by this wave
    const int half = wave >> 2;   // n-range selector
    constexpr int ntiles = NP >> 4;
    // Scalar guard: branch via SCC so EXEC is untouched (WMMA needs EXEC all-1).
    const int  ntStart = __builtin_amdgcn_readfirstlane(half * NT);
    const bool active  = (ntStart < ntiles);

    v8f acc[NT] = {};
    constexpr int sliceWords = NP * 16;            // NP*32 bf16 = NP*16 dwords
    constexpr int PER_THREAD = sliceWords / 256;   // dwords staged per thread
    static_assert(sliceWords % 256 == 0, "slice must tile over 256 threads");

    // Prologue: stage slice 0 into buffer 0.
#pragma unroll
    for (int j = 0; j < PER_THREAD; ++j)
        ((uint32_t*)Wlds)[j * 256 + tid] = Wg32[j * 256 + tid];
    __syncthreads();

    for (int kt = 0; kt < ktiles; ++kt) {
        const uint16_t* cur = Wlds + (kt & 1) * (NP * 32);
        uint16_t*       nxt = Wlds + ((kt + 1) & 1) * (NP * 32);

        // Prefetch next slice into registers (latency hidden by WMMA chain).
        uint32_t pre[PER_THREAD];
        const bool havenext = (kt + 1 < ktiles);
        if (havenext) {
#pragma unroll
            for (int j = 0; j < PER_THREAD; ++j)
                pre[j] = Wg32[(size_t)(kt + 1) * sliceWords + j * 256 + tid];
        }
        __builtin_amdgcn_sched_barrier(0);   // keep prefetch issued early

        if (active) {
            // A fragment: two contiguous 16B halves -> ds_load_b128 x2.
            ABFrag a;
            const uint16_t* Arow = A + (size_t)(mt * 16 + ln) * lda + kt * 32;
            a.q[0] = *(const uint4*)(Arow + 16 * g);       // K = 16g .. 16g+7
            a.q[1] = *(const uint4*)(Arow + 16 * g + 8);   // K = 16g+8 .. 16g+15

            // Load ALL B fragments before any WMMA.
            ABFrag b[NT];
#pragma unroll
            for (int t = 0; t < NT; ++t) {
                const uint16_t* Bp =
                    cur + (((ntStart + t) * 16 + ln) << 5) + (g << 4);
                b[t].q[0] = *(const uint4*)(Bp);           // K = 16g .. 16g+7
                b[t].q[1] = *(const uint4*)(Bp + 8);       // K = 16g+8 .. +15
            }
            __builtin_amdgcn_sched_barrier(0);  // loads above, WMMAs below

#pragma unroll
            for (int t = 0; t < NT; ++t) {
                acc[t] = __builtin_amdgcn_wmma_f32_16x16x32_bf16(
                    false, a.v, false, b[t].v, (short)0, acc[t], false, false);
            }
        }

        // Commit prefetched slice to the alternate buffer.
        if (havenext) {
#pragma unroll
            for (int j = 0; j < PER_THREAD; ++j)
                ((uint32_t*)nxt)[j * 256 + tid] = pre[j];
        }
        __syncthreads();   // nxt staged + cur fully consumed
    }

    // Epilogue: bias (+ ReLU); C layout: elem i -> row mt*16+i+8g, col nt*16+ln.
    if (active) {
#pragma unroll
        for (int t = 0; t < NT; ++t) {
            int col = (ntStart + t) * 16 + ln;
            float bcol = bias[col];
            v8f c = acc[t];
#pragma unroll
            for (int i = 0; i < 8; ++i) {
                int row = mt * 16 + i + 8 * g;
                float v = c[i] + bcol;
                if (RELU) v = v > 0.0f ? v : 0.0f;
                if (TO_LDS) {
                    Olds[row * HP + col] = f2bf(v);
                } else if (ln < OUTN) {
                    Og[(size_t)(row0 + row) * OUTN + ln] = v;
                }
            }
        }
    }
    __syncthreads();
}

// ---------------- Fused conv + 4-layer MLP ----------------
__global__ void __launch_bounds__(256)
fused_net_kernel(const float* __restrict__ x, const float* __restrict__ cw,
                 const uint16_t* __restrict__ W1p, const uint16_t* __restrict__ W2p,
                 const uint16_t* __restrict__ W3p, const uint16_t* __restrict__ W4p,
                 const float* __restrict__ b1p, const float* __restrict__ b2p,
                 const float* __restrict__ b3p, const float* __restrict__ b4p,
                 float* __restrict__ out)
{
    extern __shared__ uint16_t smem[];
    uint16_t* hA   = smem;                       // [64][704] conv out / layer-2 out
    uint16_t* hB   = smem + MTILE * K1P;         // [64][224] ping-pong
    uint16_t* Wbuf = hB + MTILE * HP;            // 2 x [224][32] weight slices

    const int tid  = threadIdx.x;
    const int row0 = blockIdx.x * MTILE;

    const float c0 = cw[0], c1 = cw[1], c2 = cw[2];
    const float c3 = cw[3], c4 = cw[4], c5 = cw[5];
    const float c6 = cw[6], c7 = cw[7], c8 = cw[8];

    // Stage 0: 3x3 valid cross-correlation, result -> bf16 in LDS.
    for (int idx = tid; idx < MTILE * CONVOUT; idx += 256) {
        int r = idx / CONVOUT;
        int p = idx - r * CONVOUT;
        int oy = p / CONVW;
        int ox = p - oy * CONVW;
        const float* xr = x + (size_t)(row0 + r) * (IMGW * IMGW) + oy * IMGW + ox;
        float s = c0 * xr[0]  + c1 * xr[1]  + c2 * xr[2]
                + c3 * xr[28] + c4 * xr[29] + c5 * xr[30]
                + c6 * xr[56] + c7 * xr[57] + c8 * xr[58];
        hA[r * K1P + p] = f2bf(s);
    }
    // Zero the K padding (cols 676..703).
    for (int idx = tid; idx < MTILE * (K1P - CONVOUT); idx += 256) {
        int r = idx / (K1P - CONVOUT);
        int c = idx - r * (K1P - CONVOUT);
        hA[r * K1P + CONVOUT + c] = 0;
    }
    __syncthreads();

    gemm_tile<HP, 7, true, true>(hA, K1P, KT1, (const uint32_t*)W1p, Wbuf, b1p,
                                 hB, nullptr, 0, tid);
    gemm_tile<HP, 7, true, true>(hB, HP, KT2, (const uint32_t*)W2p, Wbuf, b2p,
                                 hA, nullptr, 0, tid);
    gemm_tile<HP, 7, true, true>(hA, HP, KT2, (const uint32_t*)W3p, Wbuf, b3p,
                                 hB, nullptr, 0, tid);
    gemm_tile<OUTP, 1, false, false>(hB, HP, KT2, (const uint32_t*)W4p, Wbuf, b4p,
                                     nullptr, out, row0, tid);
}

// ---------------- Host entry ----------------
extern "C" void kernel_launch(void* const* d_in, const int* in_sizes, int n_in,
                              void* d_out, int out_size, void* d_ws, size_t ws_size,
                              hipStream_t stream) {
    const float* x  = (const float*)d_in[0];
    const float* cw = (const float*)d_in[1];
    const float* W1 = (const float*)d_in[2];
    const float* b1 = (const float*)d_in[3];
    const float* W2 = (const float*)d_in[4];
    const float* b2 = (const float*)d_in[5];
    const float* W3 = (const float*)d_in[6];
    const float* b3 = (const float*)d_in[7];
    const float* W4 = (const float*)d_in[8];
    const float* b4 = (const float*)d_in[9];
    float* out = (float*)d_out;

    char* ws = (char*)d_ws;
    size_t off = 0;
    auto alloc = [&](size_t bytes) -> void* {
        void* p = ws + off;
        off = (off + bytes + 255) & ~(size_t)255;
        return p;
    };
    uint16_t* W1p = (uint16_t*)alloc((size_t)KT1 * HP * 32 * 2);
    uint16_t* W2p = (uint16_t*)alloc((size_t)KT2 * HP * 32 * 2);
    uint16_t* W3p = (uint16_t*)alloc((size_t)KT2 * HP * 32 * 2);
    uint16_t* W4p = (uint16_t*)alloc((size_t)KT2 * OUTP * 32 * 2);
    float* b1p = (float*)alloc(HP * 4);
    float* b2p = (float*)alloc(HP * 4);
    float* b3p = (float*)alloc(HP * 4);
    float* b4p = (float*)alloc(OUTP * 4);

    auto blocks = [](int total) { return (total + 255) / 256; };

    pack_w_kernel<<<blocks(KT1 * HP * 32), 256, 0, stream>>>(W1, CONVOUT, HID, W1p, KT1, HP);
    pack_w_kernel<<<blocks(KT2 * HP * 32), 256, 0, stream>>>(W2, HID, HID, W2p, KT2, HP);
    pack_w_kernel<<<blocks(KT2 * HP * 32), 256, 0, stream>>>(W3, HID, HID, W3p, KT2, HP);
    pack_w_kernel<<<blocks(KT2 * OUTP * 32), 256, 0, stream>>>(W4, HID, OUTN, W4p, KT2, OUTP);
    pack_b_kernel<<<1, 256, 0, stream>>>(b1, HID, b1p, HP);
    pack_b_kernel<<<1, 256, 0, stream>>>(b2, HID, b2p, HP);
    pack_b_kernel<<<1, 256, 0, stream>>>(b3, HID, b3p, HP);
    pack_b_kernel<<<1, 256, 0, stream>>>(b4, OUTN, b4p, OUTP);

    fused_net_kernel<<<BATCH / MTILE, 256, SMEM_BYTES, stream>>>(
        x, cw, W1p, W2p, W3p, W4p, b1p, b2p, b3p, b4p, out);
}
